// GIN_4layer_66571993088585
// MI455X (gfx1250) — compile-verified, matrix operand order checked by reading, not run on hardware
//
#include <hip/hip_runtime.h>
#include <hip/hip_bf16.h>

#define NN     50000
#define EE     800000
#define FIN    128
#define HID    256
#define GG     64
#define NGENE  250
#define BN_EPS 1e-5f

typedef __bf16 bf16_t;
typedef __attribute__((ext_vector_type(16))) __bf16 v16bf;
typedef __attribute__((ext_vector_type(8)))  __bf16 v8bf;
typedef __attribute__((ext_vector_type(2)))  __bf16 v2bf;
typedef __attribute__((ext_vector_type(8)))  float  v8f;

#if __has_builtin(__builtin_amdgcn_cvt_pk_bf16_f32)
#define HAVE_PKBF 1
#else
#define HAVE_PKBF 0
#endif

static __device__ __forceinline__ bf16_t f2bf(float f) {
  unsigned u = __builtin_bit_cast(unsigned, f);
  u = (u + 0x7fffu + ((u >> 16) & 1u)) >> 16;   // round-to-nearest-even
  unsigned short s = (unsigned short)u;
  return __builtin_bit_cast(bf16_t, s);
}

static __device__ __forceinline__ v2bf pk2bf(float a, float b) {
#if HAVE_PKBF
  return __builtin_amdgcn_cvt_pk_bf16_f32(a, b);   // v_cvt_pk_bf16_f32 (RNE)
#else
  v2bf r; r[0] = f2bf(a); r[1] = f2bf(b); return r;
#endif
}

// generic pointer (known to point into LDS) -> 32-bit LDS byte offset
typedef __attribute__((address_space(3))) const void* lds_cvp;
static __device__ __forceinline__ unsigned ldsoff(const void* p) {
  return (unsigned)(unsigned long long)(lds_cvp)p;
}

// CDNA5 async global->LDS copy (bypasses VGPRs, tracked by ASYNCcnt)
static __device__ __forceinline__ void async_ld_b128(unsigned lds, const void* g) {
  asm volatile("global_load_async_to_lds_b128 %0, %1, off" :: "v"(lds), "v"(g) : "memory");
}
static __device__ __forceinline__ void wait_async0() {
  asm volatile("s_wait_asynccnt 0x0" ::: "memory");
}

// build a 16x32 bf16 A-fragment row slice from an f32 row pointer
static __device__ __forceinline__ v16bf mk_afrag_f32(const float* zr) {
  const float4 f0 = *(const float4*)(zr);
  const float4 f1 = *(const float4*)(zr + 4);
  const float4 f2 = *(const float4*)(zr + 16);
  const float4 f3 = *(const float4*)(zr + 20);
  v16bf a;
  v2bf* ap = (v2bf*)&a;
  ap[0] = pk2bf(f0.x, f0.y); ap[1] = pk2bf(f0.z, f0.w);
  ap[2] = pk2bf(f1.x, f1.y); ap[3] = pk2bf(f1.z, f1.w);
  ap[4] = pk2bf(f2.x, f2.y); ap[5] = pk2bf(f2.z, f2.w);
  ap[6] = pk2bf(f3.x, f3.y); ap[7] = pk2bf(f3.z, f3.w);
  return a;
}

// ---------------- small utility kernels ----------------

__global__ void kzero(float* p, long long n) {
  long long i = blockIdx.x * (long long)blockDim.x + threadIdx.x;
  long long stride = (long long)gridDim.x * blockDim.x;
  for (; i < n; i += stride) p[i] = 0.0f;
}

__global__ void kcopy4(float4* d, const float4* s, long long n4) {
  long long i = blockIdx.x * (long long)blockDim.x + threadIdx.x;
  if (i < n4) d[i] = s[i];
}

__global__ void kcount(const int* batch, int* cnt, int n) {
  int i = blockIdx.x * blockDim.x + threadIdx.x;
  if (i < n) atomicAdd(&cnt[batch[i]], 1);
}

// one-time per launch: W [K][256] f32 -> WT [256][K] bf16 (transposed = LDS layout)
__global__ void kconvw(const float* __restrict__ W, bf16_t* __restrict__ WT, int K) {
  int i = blockIdx.x * blockDim.x + threadIdx.x;
  if (i < K * 256) {
    int k = i >> 8, n = i & 255;
    WT[n * K + k] = f2bf(W[i]);
  }
}

// z[dst] += h[src]  (z pre-initialized with h => z = h + segment_sum(h[src], dst)).
// Atomics resolve in the 192MB L2 (h is 51MB -> fully L2-resident).
template<int D>
__global__ void kscatter(float* z, const float* __restrict__ h,
                         const int* __restrict__ src, const int* __restrict__ dst) {
  long long tid = blockIdx.x * (long long)blockDim.x + threadIdx.x;
  constexpr int Q = D >> 2;
  long long total = (long long)EE * Q;
  if (tid >= total) return;
  int e = (int)(tid / Q);
  int c = (int)(tid % Q) << 2;
  int s = src[e], d = dst[e];
  const float4 v = *(const float4*)(h + (long long)s * D + c);
  float* p = z + (long long)d * D + c;
  atomicAdd(p + 0, v.x); atomicAdd(p + 1, v.y);
  atomicAdd(p + 2, v.z); atomicAdd(p + 3, v.w);
}

// ---------------- fused GIN MLP: z2 = relu(z @ W1 + b1) @ W2 + b2 ----------------
// WT = pre-converted bf16 [W1^T | W2^T] contiguous; staged into LDS via async
// global->LDS b128 copies. 320KB WGP LDS holds both weight matrices (up to 256KB)
// plus a per-wave 16x256 bf16 GEMM1->GEMM2 staging tile. Each block computes 256
// node-rows (4 waves x 4 strips of 16), amortizing the one-time LDS fill.
template<int K>
__launch_bounds__(128, 1)
__global__ void kgin_mlp(const float* __restrict__ Z, const bf16_t* __restrict__ WT,
                         const float* __restrict__ B1, const float* __restrict__ B2,
                         float* __restrict__ Z2, int nrows) {
  extern __shared__ char smem[];
  bf16_t* w1t = (bf16_t*)smem;            // [256][K]   W1 transposed
  bf16_t* w2t = w1t + 256 * K;            // [256][256] W2 transposed
  bf16_t* z1  = w2t + 256 * 256;          // [4 waves][16][256] staging

  const int tid  = threadIdx.x;           // 128 threads = 4 waves (wave32)
  const int lane = tid & 31;
  const int wv   = tid >> 5;
  const int hi   = lane >> 4;             // which K-half of the fragment this lane holds
  const int l15  = lane & 15;

  // Stage both (pre-transposed, pre-bf16) weight matrices with async copies.
  constexpr unsigned WBYTES = (256u * K + 256u * 256u) * 2u;
  {
    const unsigned lbase = ldsoff(w1t);
    const char* g = (const char*)WT;
    for (unsigned off = (unsigned)tid * 16u; off < WBYTES; off += 128u * 16u)
      async_ld_b128(lbase + off, g + off);
    wait_async0();
  }
  __syncthreads();

  bf16_t* z1w = z1 + wv * 16 * 256;
  constexpr int KS1 = K / 32;

  for (int s = 0; s < 4; ++s) {
    const int row0 = blockIdx.x * 256 + s * 64 + wv * 16;   // wave's 16-row strip
    if (row0 >= nrows) break;                               // wave-uniform exit
    int arow = row0 + l15; if (arow > nrows - 1) arow = nrows - 1;

    if (s < 3) {                                            // prefetch next strip's rows
      int prow = row0 + 64 + l15; if (prow > nrows - 1) prow = nrows - 1;
      __builtin_prefetch((const void*)(Z + (long long)prow * K), 0, 1);
    }

    // A fragments for GEMM1 (ISA 16-bit A layout: lane holds row M=l15,
    // K = {0..7,16..23} / {8..15,24..31} chosen by lane half, per 32-K step).
    v16bf afr[KS1];
    #pragma unroll
    for (int kk = 0; kk < KS1; ++kk)
      afr[kk] = mk_afrag_f32(Z + (long long)arow * K + kk * 32 + hi * 8);

    // GEMM1: 16 column tiles; B fragment column n = contiguous row of w1t.
    for (int nt = 0; nt < 16; ++nt) {
      const int n = nt * 16 + l15;
      const float bias = B1[n];
      v8f c;
      #pragma unroll
      for (int i = 0; i < 8; ++i) c[i] = bias;
      const bf16_t* wc = w1t + n * K;
      #pragma unroll
      for (int kk = 0; kk < KS1; ++kk) {
        const bf16_t* br = wc + kk * 32 + hi * 8;
        v8bf blo = *(const v8bf*)(br);
        v8bf bhi = *(const v8bf*)(br + 16);
        v16bf b;
        #pragma unroll
        for (int i = 0; i < 8; ++i) { b[i] = blo[i]; b[i + 8] = bhi[i]; }
        c = __builtin_amdgcn_wmma_f32_16x16x32_bf16(false, afr[kk], false, b,
                                                    (short)0, c, false, false);
      }
      // D layout: col n = l15(+nt*16), rows M = hi*8 + j. ReLU + packed-bf16 restage.
      #pragma unroll
      for (int j = 0; j < 8; j += 2) {
        v2bf p = pk2bf(fmaxf(c[j], 0.0f), fmaxf(c[j + 1], 0.0f));
        z1w[(hi * 8 + j)     * 256 + n] = p[0];
        z1w[(hi * 8 + j + 1) * 256 + n] = p[1];
      }
    }
    // same-wave LDS RAW: compiler inserts s_wait_dscnt; no cross-wave sharing -> no barrier

    // GEMM2 A fragments from the staged z1 tile (bf16, row-major).
    v16bf afr2[8];
    #pragma unroll
    for (int kk = 0; kk < 8; ++kk) {
      const bf16_t* zr = z1w + l15 * 256 + kk * 32 + hi * 8;
      v8bf alo = *(const v8bf*)(zr);
      v8bf ahi = *(const v8bf*)(zr + 16);
      v16bf a;
      #pragma unroll
      for (int i = 0; i < 8; ++i) { a[i] = alo[i]; a[i + 8] = ahi[i]; }
      afr2[kk] = a;
    }

    // GEMM2: z2 = z1 @ W2 + b2, streamed to global f32.
    for (int nt = 0; nt < 16; ++nt) {
      const int n = nt * 16 + l15;
      const float bias = B2[n];
      v8f c;
      #pragma unroll
      for (int i = 0; i < 8; ++i) c[i] = bias;
      const bf16_t* wc = w2t + n * 256;
      #pragma unroll
      for (int kk = 0; kk < 8; ++kk) {
        const bf16_t* br = wc + kk * 32 + hi * 8;
        v8bf blo = *(const v8bf*)(br);
        v8bf bhi = *(const v8bf*)(br + 16);
        v16bf b;
        #pragma unroll
        for (int i = 0; i < 8; ++i) { b[i] = blo[i]; b[i + 8] = bhi[i]; }
        c = __builtin_amdgcn_wmma_f32_16x16x32_bf16(false, afr2[kk], false, b,
                                                    (short)0, c, false, false);
      }
      #pragma unroll
      for (int j = 0; j < 8; ++j) {
        int r = row0 + hi * 8 + j;
        if (r < nrows) Z2[(long long)r * 256 + n] = c[j];
      }
    }
  }
}

// ---------------- BatchNorm statistics / apply / pooling ----------------

__global__ void kcolstats(const float* __restrict__ z2, float* stats, int n) {
  int col = threadIdx.x;                 // 256 threads, one column each (coalesced rows)
  int r0 = blockIdx.x * 256;
  int r1 = r0 + 256; if (r1 > n) r1 = n;
  float s = 0.f, ss = 0.f;
  for (int r = r0; r < r1; ++r) {
    float v = z2[(long long)r * 256 + col];
    s += v; ss += v * v;
  }
  atomicAdd(&stats[col], s);
  atomicAdd(&stats[256 + col], ss);
}

__global__ void kbnfin(const float* stats, float* murs, int n) {
  int c = threadIdx.x;
  float mu  = stats[c] / (float)n;
  float var = stats[256 + c] / (float)n - mu * mu;
  murs[c]       = mu;
  murs[256 + c] = rsqrtf(fmaxf(var, 0.0f) + BN_EPS);
}

// h = relu(BN(z2)); pool sums + maxes per graph (post-ReLU values >= 0, so
// atomicMax on raw float bits is order-preserving and init=0 is correct).
__global__ void kbnrp(const float* __restrict__ z2, const float* __restrict__ murs,
                      const float* __restrict__ gam, const float* __restrict__ bet,
                      float* __restrict__ h, const int* __restrict__ batch,
                      float* psum, unsigned int* pmax, int n) {
  long long tid = blockIdx.x * (long long)blockDim.x + threadIdx.x;
  if (tid >= (long long)n * 256) return;
  int r = (int)(tid >> 8), c = (int)(tid & 255);
  float v = z2[tid];
  v = gam[c] * (v - murs[c]) * murs[256 + c] + bet[c];
  v = fmaxf(v, 0.0f);
  h[tid] = v;
  int g = batch[r];
  atomicAdd(&psum[g * 256 + c], v);
  atomicMax(&pmax[g * 256 + c], __float_as_uint(v));
}

__global__ void kpoolacc(float* acc, const unsigned int* pmax, const float* psum, const int* cnt) {
  int tid = blockIdx.x * blockDim.x + threadIdx.x;
  if (tid >= GG * 512) return;
  int g = tid >> 9, c = tid & 511;
  float v = (c < 256) ? __uint_as_float(pmax[g * 256 + c])
                      : psum[g * 256 + (c - 256)] / fmaxf((float)cnt[g], 1.0f);
  acc[tid] += v;
}

// Tiny readout GEMMs: 64xK @ KxN, ~25 MFLOP total -> plain VALU is fine.
__global__ void klin(const float* __restrict__ A, const float* __restrict__ W,
                     const float* __restrict__ b, float* out,
                     int M, int Kd, int Nd, int relu) {
  int tid = blockIdx.x * blockDim.x + threadIdx.x;
  if (tid >= M * Nd) return;
  int m = tid / Nd, j = tid % Nd;
  const float* a = A + (long long)m * Kd;
  float s = b[j];
  for (int k = 0; k < Kd; ++k) s = fmaf(a[k], W[(long long)k * Nd + j], s);
  if (relu) s = fmaxf(s, 0.0f);
  out[tid] = s;
}

// ---------------- orchestration ----------------

static inline dim3 gl(long long n, int b) { return dim3((unsigned)((n + b - 1) / b)); }

extern "C" void kernel_launch(void* const* d_in, const int* in_sizes, int n_in,
                              void* d_out, int out_size, void* d_ws, size_t ws_size,
                              hipStream_t stream) {
  const float* x     = (const float*)d_in[0];
  const int*   ei    = (const int*)d_in[1];
  const int*   batch = (const int*)d_in[2];
  const float* w1a   = (const float*)d_in[3];
  const float* b1a   = (const float*)d_in[4];
  const float* w1b   = (const float*)d_in[5];
  const float* b1b   = (const float*)d_in[6];
  const float* W_in  = (const float*)d_in[7];
  const float* b_in  = (const float*)d_in[8];
  const float* W_out = (const float*)d_in[9];
  const float* b_out = (const float*)d_in[10];
  const float* gamma = (const float*)d_in[11];
  const float* beta  = (const float*)d_in[12];
  const float* l1w   = (const float*)d_in[13];
  const float* l1b   = (const float*)d_in[14];
  const float* l2w   = (const float*)d_in[15];
  const float* l2b   = (const float*)d_in[16];
  const float* l3w   = (const float*)d_in[17];
  const float* l3b   = (const float*)d_in[18];
  const int* src = ei, * dst = ei + EE;

  float* ws = (float*)d_ws;
  float* h     = ws;                             // [N,256]
  float* z     = h  + (size_t)NN * HID;          // [N,256]  (h + agg)
  float* z2    = z  + (size_t)NN * HID;          // [N,256]  (MLP out, pre-BN)
  float* stats = z2 + (size_t)NN * HID;          // [512] sum|sumsq
  float* murs  = stats + 512;                    // [512] mu|rstd
  float* psum  = murs + 512;                     // [G,256]
  unsigned int* pmax = (unsigned int*)(psum + GG * HID); // [G,256]
  int*   cnt   = (int*)(pmax + GG * HID);        // [G]
  float* acc   = (float*)(cnt + GG);             // [G,512]
  float* z1h   = acc + GG * 512;                 // [G,256]
  float* z2h   = z1h + GG * HID;                 // [G,128]
  // pre-converted bf16 transposed weights, per-layer contiguous [W1T | W2T]
  bf16_t* wt0 = (bf16_t*)(z2h + GG * 128);                 // layer0: 256*128 + 256*256
  bf16_t* wtL = wt0 + (256 * FIN + 256 * 256);             // layers 1..3: 2*256*256 each

  // one-time (per call) weight convert+transpose to bf16
  kconvw<<<gl(FIN * 256, 256), 256, 0, stream>>>(w1a, wt0, FIN);
  kconvw<<<gl(HID * 256, 256), 256, 0, stream>>>(w1b, wt0 + 256 * FIN, HID);
  for (int i = 0; i < 3; ++i) {
    bf16_t* wi = wtL + (size_t)i * 2 * 256 * 256;
    kconvw<<<gl(HID * 256, 256), 256, 0, stream>>>(W_in  + (size_t)i * HID * HID, wi, HID);
    kconvw<<<gl(HID * 256, 256), 256, 0, stream>>>(W_out + (size_t)i * HID * HID, wi + 256 * 256, HID);
  }

  // zero accumulators every call (deterministic; harness replays the graph)
  kzero<<<gl(GG * 512, 256), 256, 0, stream>>>(acc, GG * 512);
  kzero<<<1, 64, 0, stream>>>((float*)cnt, GG);
  kcount<<<gl(NN, 256), 256, 0, stream>>>(batch, cnt, NN);

  auto layer = [&](const float* in, int K, const bf16_t* WT, const float* B1,
                   const float* B2, const float* gam, const float* bet) {
    long long n4 = (long long)NN * K / 4;
    kcopy4<<<gl(n4, 256), 256, 0, stream>>>((float4*)z, (const float4*)in, n4);
    if (K == 128)
      kscatter<128><<<gl((long long)EE * 32, 256), 256, 0, stream>>>(z, in, src, dst);
    else
      kscatter<256><<<gl((long long)EE * 64, 256), 256, 0, stream>>>(z, in, src, dst);
    kzero<<<1, 512, 0, stream>>>(stats, 512);
    unsigned blocks = (NN + 255) / 256;
    if (K == 128)
      kgin_mlp<128><<<blocks, 128, (256 * 128 + 256 * 256 + 64 * 256) * 2, stream>>>(
          z, WT, B1, B2, z2, NN);
    else
      kgin_mlp<256><<<blocks, 128, (256 * 256 + 256 * 256 + 64 * 256) * 2, stream>>>(
          z, WT, B1, B2, z2, NN);
    kcolstats<<<gl(NN, 256), 256, 0, stream>>>(z2, stats, NN);
    kbnfin<<<1, 256, 0, stream>>>(stats, murs, NN);
    kzero<<<gl(GG * HID, 256), 256, 0, stream>>>(psum, GG * HID);
    kzero<<<gl(GG * HID, 256), 256, 0, stream>>>((float*)pmax, GG * HID);
    kbnrp<<<gl((long long)NN * 256, 256), 256, 0, stream>>>(
        z2, murs, gam, bet, h, batch, psum, pmax, NN);
    kpoolacc<<<gl(GG * 512, 256), 256, 0, stream>>>(acc, pmax, psum, cnt);
  };

  layer(x, 128, wt0, b1a, b1b, gamma, beta);
  for (int i = 0; i < 3; ++i)
    layer(h, 256, wtL + (size_t)i * 2 * 256 * 256, b_in + i * HID, b_out + i * HID,
          gamma + (i + 1) * HID, beta + (i + 1) * HID);

  klin<<<gl(GG * 256, 256), 256, 0, stream>>>(acc, l1w, l1b, z1h, GG, 512, 256, 1);
  klin<<<gl(GG * 128, 128), 128, 0, stream>>>(z1h, l2w, l2b, z2h, GG, 256, 128, 1);
  klin<<<gl(GG * NGENE, 256), 256, 0, stream>>>(z2h, l3w, l3b, (float*)d_out, GG, 128, NGENE, 0);

  (void)in_sizes; (void)n_in; (void)out_size; (void)ws_size;
}
